// CAN_81982335746243
// MI455X (gfx1250) — matrix operand order repaired and checked by reading, not compiled, not run on hardware
//
#include <hip/hip_runtime.h>
#include <hip/hip_bf16.h>

typedef __attribute__((ext_vector_type(16))) _Float16 v16h;
typedef __attribute__((ext_vector_type(8)))  _Float16 v8h;
typedef __attribute__((ext_vector_type(8)))  float    v8f;

#define NE 4096      // edges / 1-cells
#define NN 2048      // nodes
#define C0 64
#define DH 128       // hidden = HEADS*OUT
#define KEEP 2048
#define EPS_SKIP 1.000001f

// ---------------- WMMA fragment helpers (wave32 layouts, ISA 7.12.2) -----------------

// A fragment: rows [row0..row0+15], K block [k0..k0+31], row-major f16, ld=lda (mult of 8)
__device__ __forceinline__ v16h load_A(const _Float16* __restrict__ A, int lda,
                                       int row0, int k0, int lane) {
    int r  = row0 + (lane & 15);
    int kb = (lane < 16) ? 0 : 16;
    const _Float16* p = A + (size_t)r * lda + k0 + kb;
    v8h lo = *(const v8h*)p;          // K = k0+kb .. +7
    v8h hi = *(const v8h*)(p + 8);    // K = k0+kb+8 .. +15
    return __builtin_shufflevector(lo, hi, 0, 1, 2, 3, 4, 5, 6, 7,
                                           8, 9, 10, 11, 12, 13, 14, 15);
}

// B fragment from TRANSPOSED B (layout [N, K] row-major): 16 contiguous halfs per lane
__device__ __forceinline__ v16h load_BT(const _Float16* __restrict__ BT, int ldk,
                                        int n0, int k0, int lane) {
    int c   = n0 + (lane & 15);
    int kbb = (lane < 16) ? 0 : 16;
    return *(const v16h*)(BT + (size_t)c * ldk + k0 + kbb);
}

// ---------------- kernels ------------------------------------------------------------

__global__ void k_cvt_f16(const float* __restrict__ in, _Float16* __restrict__ out, int n) {
    int i = blockIdx.x * blockDim.x + threadIdx.x;
    if (i < n) out[i] = (_Float16)in[i];
}

// tiled transpose f32[R,C] -> f16[C,R]  (R, C multiples of 16; block = 256 threads)
__global__ void k_t16(const float* __restrict__ in, _Float16* __restrict__ out,
                      int R, int C) {
    __shared__ _Float16 tile[16][17];
    int rb = blockIdx.x * 16, cb = blockIdx.y * 16;
    int tx = threadIdx.x & 15, ty = threadIdx.x >> 4;
    tile[ty][tx] = (_Float16)in[(size_t)(rb + ty) * C + cb + tx];
    __syncthreads();
    out[(size_t)(cb + ty) * R + rb + tx] = tile[tx][ty];
}

// lift: X[e,0:64] = relu(concat(x0[src],x0[tgt]) @ Wlift) ; one wave per 16 edges
// WliftT is [64, 128] f16 (transposed)
__global__ void k_lift(const float* __restrict__ x0, const int* __restrict__ eidx,
                       const _Float16* __restrict__ WliftT, float* __restrict__ X) {
    int lane = threadIdx.x;
    int s    = eidx[blockIdx.x * 16 + (lane & 15)];
    int t    = eidx[NE + blockIdx.x * 16 + (lane & 15)];
    int kb   = (lane < 16) ? 0 : 16;
    v8f zero = {};
    v8f acc[4] = {zero, zero, zero, zero};
    for (int k0 = 0; k0 < 128; k0 += 32) {
        int kA = k0 + kb, kB = k0 + kb + 8;    // each 8-seg lies wholly in one half
        const float* pA = (kA < C0) ? x0 + (size_t)s * C0 + kA : x0 + (size_t)t * C0 + kA - C0;
        const float* pB = (kB < C0) ? x0 + (size_t)s * C0 + kB : x0 + (size_t)t * C0 + kB - C0;
        v8f fA = *(const v8f*)pA;
        v8f fB = *(const v8f*)pB;
        v16h a;
#pragma unroll
        for (int v = 0; v < 8; ++v) { a[v] = (_Float16)fA[v]; a[8 + v] = (_Float16)fB[v]; }
#pragma unroll
        for (int nt = 0; nt < 4; ++nt) {
            v16h b = load_BT(WliftT, 128, nt * 16, k0, lane);
            acc[nt] = __builtin_amdgcn_wmma_f32_16x16x32_f16(false, a, false, b,
                                                             (short)0, acc[nt], false, false);
        }
    }
    int mb = (lane < 16) ? 0 : 8;
    int c  = lane & 15;
#pragma unroll
    for (int nt = 0; nt < 4; ++nt)
#pragma unroll
        for (int v = 0; v < 8; ++v) {
            float val = acc[nt][v];
            X[(size_t)(blockIdx.x * 16 + mb + v) * DH + nt * 16 + c] = val > 0.f ? val : 0.f;
        }
}

__global__ void k_copy_x1(const float* __restrict__ x1, float* __restrict__ X) {
    int i = blockIdx.x * blockDim.x + threadIdx.x;
    if (i < NE * C0) { int e = i >> 6, c = i & 63; X[(size_t)e * DH + C0 + c] = x1[i]; }
}

// C[M,N] = Ah[M,K] @ BT[N,K]^T ; one wave per 16x16 tile
__global__ void k_gemm(const _Float16* __restrict__ Ah, const _Float16* __restrict__ BT,
                       float* __restrict__ C, int M, int N, int K) {
    int lane = threadIdx.x;
    int tm = blockIdx.x, tn = blockIdx.y;
    v8f acc = {};
    for (int k0 = 0; k0 < K; k0 += 32) {
        v16h a = load_A(Ah, K, tm * 16, k0, lane);
        v16h b = load_BT(BT, K, tn * 16, k0, lane);
        acc = __builtin_amdgcn_wmma_f32_16x16x32_f16(false, a, false, b,
                                                     (short)0, acc, false, false);
    }
    int mb = (lane < 16) ? 0 : 8;
    int c  = tn * 16 + (lane & 15);
#pragma unroll
    for (int v = 0; v < 8; ++v) C[(size_t)(tm * 16 + mb + v) * N + c] = acc[v];
}

// per-row head logits (separated per-head src arrays for coalesced reuse)
__global__ void k_logits(const float* __restrict__ XM, const float* __restrict__ a_src,
                         const float* __restrict__ a_dst, float* __restrict__ AS0,
                         float* __restrict__ AS1, float* __restrict__ AD) {
    int n = blockIdx.x * blockDim.x + threadIdx.x;
    if (n >= NE) return;
    float s[2], d[2];
#pragma unroll
    for (int h = 0; h < 2; ++h) {
        float ss = 0.f, dd = 0.f;
        for (int k = 0; k < 64; ++k) {
            float v = XM[(size_t)n * DH + h * 64 + k];
            ss += v * a_src[h * 64 + k];
            dd += v * a_dst[h * 64 + k];
        }
        s[h] = ss; d[h] = dd;
    }
    AS0[n] = s[0]; AS1[n] = s[1];
    AD[n * 2 + 0] = d[0]; AD[n * 2 + 1] = d[1];
}

// masked row-max of a_s, folded through leakyrelu with a_d -> exact softmax max
__global__ void k_rowmax(const float* __restrict__ mask, const float* __restrict__ AS0,
                         const float* __restrict__ AS1, const float* __restrict__ AD,
                         float* __restrict__ MR) {
    int i = blockIdx.x, tid = threadIdx.x;
    float m0 = -3.4e38f, m1 = -3.4e38f;
    for (int j = tid; j < NE; j += 256) {
        bool nb = (mask[(size_t)i * NE + j] != 0.f) || (j == i);
        if (nb) {
            m0 = fmaxf(m0, AS0[j]);
            m1 = fmaxf(m1, AS1[j]);
        }
    }
    __shared__ float s0[256], s1[256];
    s0[tid] = m0; s1[tid] = m1; __syncthreads();
    for (int off = 128; off > 0; off >>= 1) {
        if (tid < off) { s0[tid] = fmaxf(s0[tid], s0[tid + off]);
                         s1[tid] = fmaxf(s1[tid], s1[tid + off]); }
        __syncthreads();
    }
    if (tid == 0) {
        float l0 = AD[i * 2 + 0] + s0[0]; l0 = l0 > 0.f ? l0 : 0.2f * l0;
        float l1 = AD[i * 2 + 1] + s1[0]; l1 = l1 > 0.f ? l1 : 0.2f * l1;
        MR[i * 2 + 0] = l0; MR[i * 2 + 1] = l1;
    }
}

// fused masked softmax + aggregation with transposed XM (f16 [DH, NE]):
// OUT[i,h*64+d] = sum_j p_ijh * XM[j,h*64+d] / sum_j p_ijh
// one wave per 16 target rows; 8 WMMAs per 32-source chunk (2 heads x 4 N-tiles)
__global__ void __launch_bounds__(32)
k_attn(const float* __restrict__ mask, const float* __restrict__ AS0,
       const float* __restrict__ AS1, const float* __restrict__ AD,
       const float* __restrict__ MR, const _Float16* __restrict__ XMT,
       float* __restrict__ OUTA) {
    __shared__ float sAS0[NE];    // 16 KB per head: source logits, reused by every chunk
    __shared__ float sAS1[NE];
    __shared__ float sden[16][2];
    int lane = threadIdx.x;
    for (int tix = lane; tix < NE; tix += 32) { sAS0[tix] = AS0[tix]; sAS1[tix] = AS1[tix]; }
    __syncthreads();

    int i0 = blockIdx.x * 16;
    int r  = i0 + (lane & 15);
    const float* mrow = mask + (size_t)r * NE;
    float ad0 = AD[r * 2 + 0], ad1 = AD[r * 2 + 1];
    float mr0 = MR[r * 2 + 0], mr1 = MR[r * 2 + 1];
    int kb  = (lane < 16) ? 0 : 16;
    int kbb = (lane < 16) ? 0 : 16;
    int cc  = lane & 15;

    v8f zero = {};
    v8f acc[2][4] = {{zero, zero, zero, zero}, {zero, zero, zero, zero}};
    float ds0 = 0.f, ds1 = 0.f;

    for (int j0 = 0; j0 < NE; j0 += 32) {
        __builtin_prefetch(mrow + j0 + 32 + kb, 0, 0);
        v8f mkA = *(const v8f*)(mrow + j0 + kb);          // 32B vector mask loads
        v8f mkB = *(const v8f*)(mrow + j0 + kb + 8);
        v8f s0A = *(const v8f*)&sAS0[j0 + kb];
        v8f s0B = *(const v8f*)&sAS0[j0 + kb + 8];
        v8f s1A = *(const v8f*)&sAS1[j0 + kb];
        v8f s1B = *(const v8f*)&sAS1[j0 + kb + 8];
        v16h a0, a1;
#pragma unroll
        for (int v = 0; v < 16; ++v) {
            int j    = j0 + kb + v;
            float mk = (v < 8) ? mkA[v & 7] : mkB[v & 7];
            float s0 = (v < 8) ? s0A[v & 7] : s0B[v & 7];
            float s1 = (v < 8) ? s1A[v & 7] : s1B[v & 7];
            bool nb  = (mk != 0.f) || (j == r);
            float l0 = ad0 + s0; l0 = l0 > 0.f ? l0 : 0.2f * l0;
            float l1 = ad1 + s1; l1 = l1 > 0.f ? l1 : 0.2f * l1;
            float p0 = nb ? __expf(l0 - mr0) : 0.f;
            float p1 = nb ? __expf(l1 - mr1) : 0.f;
            ds0 += p0; ds1 += p1;
            a0[v] = (_Float16)p0; a1[v] = (_Float16)p1;
        }
#pragma unroll
        for (int nt = 0; nt < 4; ++nt) {
            v16h b0 = *(const v16h*)(XMT + (size_t)(nt * 16 + cc) * NE + j0 + kbb);
            v16h b1 = *(const v16h*)(XMT + (size_t)(64 + nt * 16 + cc) * NE + j0 + kbb);
            acc[0][nt] = __builtin_amdgcn_wmma_f32_16x16x32_f16(false, a0, false, b0,
                                                                (short)0, acc[0][nt], false, false);
            acc[1][nt] = __builtin_amdgcn_wmma_f32_16x16x32_f16(false, a1, false, b1,
                                                                (short)0, acc[1][nt], false, false);
        }
    }
    // row denominators: lane L holds half the K range for row L&15; combine with L^16
    ds0 += __shfl_xor(ds0, 16, 32);
    ds1 += __shfl_xor(ds1, 16, 32);
    if (lane < 16) { sden[lane][0] = ds0; sden[lane][1] = ds1; }
    __syncthreads();

    int mb = (lane < 16) ? 0 : 8;
#pragma unroll
    for (int nt = 0; nt < 4; ++nt)
#pragma unroll
        for (int v = 0; v < 8; ++v) {
            int m = mb + v;
            OUTA[(size_t)(i0 + m) * DH + nt * 16 + cc]      = acc[0][nt][v] / sden[m][0];
            OUTA[(size_t)(i0 + m) * DH + 64 + nt * 16 + cc] = acc[1][nt][v] / sden[m][1];
        }
}

__global__ void k_combine(const float* __restrict__ L, const float* __restrict__ U,
                          const float* __restrict__ S, float* __restrict__ O, int n) {
    int i = blockIdx.x * blockDim.x + threadIdx.x;
    if (i < n) {
        float v = L[i] + U[i] + EPS_SKIP * S[i];
        O[i] = v > 0.f ? v : 0.f;
    }
}

__global__ void k_pool(const float* __restrict__ X, const float* __restrict__ ap,
                       float* __restrict__ Z) {
    int e = blockIdx.x * blockDim.x + threadIdx.x;
    if (e < NE) {
        float s = 0.f;
        for (int k = 0; k < DH; ++k) s += X[(size_t)e * DH + k] * ap[k];
        Z[e] = s;
    }
}

// exact top-k by rank counting (stable: ties broken by lower index, matches lax.top_k)
__global__ void k_topk(const float* __restrict__ Z, const float* __restrict__ X,
                       float* __restrict__ out) {
    int e = blockIdx.x, tid = threadIdx.x;   // 128 threads
    float ze = Z[e];
    int cnt = 0;
    for (int j = tid; j < NE; j += 128) {
        float zj = Z[j];
        if (zj > ze || (zj == ze && j < e)) cnt++;
    }
    __shared__ int s[128];
    s[tid] = cnt; __syncthreads();
    for (int off = 64; off > 0; off >>= 1) {
        if (tid < off) s[tid] += s[tid + off];
        __syncthreads();
    }
    int rank = s[0];
    if (rank < KEEP) {
        float g = 1.f / (1.f + __expf(-ze));
        out[(size_t)rank * DH + tid] = X[(size_t)e * DH + tid] * g;
    }
}

// ---------------- host orchestration -------------------------------------------------

extern "C" void kernel_launch(void* const* d_in, const int* in_sizes, int n_in,
                              void* d_out, int out_size, void* d_ws, size_t ws_size,
                              hipStream_t stream) {
    const float* x_0   = (const float*)d_in[0];
    const float* x_1   = (const float*)d_in[1];
    const int*   eidx  = (const int*)  d_in[2];
    const float* mL    = (const float*)d_in[3];
    const float* mU    = (const float*)d_in[4];
    const float* Wlift = (const float*)d_in[5];
    const float* W[2][3]  = {{(const float*)d_in[6],  (const float*)d_in[9],  (const float*)d_in[12]},
                             {(const float*)d_in[13], (const float*)d_in[16], (const float*)d_in[19]}};
    const float* aS[2][2] = {{(const float*)d_in[7],  (const float*)d_in[10]},
                             {(const float*)d_in[14], (const float*)d_in[17]}};
    const float* aD[2][2] = {{(const float*)d_in[8],  (const float*)d_in[11]},
                             {(const float*)d_in[15], (const float*)d_in[18]}};
    const float* att_pool = (const float*)d_in[20];
    float* out = (float*)d_out;

    char* ws = (char*)d_ws;
    constexpr size_t NF  = (size_t)NE * DH;                 // 524288
    constexpr size_t SZF = NF * sizeof(float);              // 2 MB
    constexpr size_t SZH = NF * sizeof(_Float16);           // 1 MB
    constexpr size_t SZV = (size_t)NE * sizeof(float);      // 16 KB
    size_t off = 0;
    auto alloc = [&](size_t b) { size_t o = off; off += (b + 255) & ~(size_t)255; return o; };
    float*     XA   = (float*)(ws + alloc(SZF));
    float*     XB   = (float*)(ws + alloc(SZF));
    _Float16*  XH   = (_Float16*)(ws + alloc(SZH));
    float*     XML  = (float*)(ws + alloc(SZF));
    float*     XMU  = (float*)(ws + alloc(SZF));
    _Float16*  XMLT = (_Float16*)(ws + alloc(SZH));  // transposed [DH, NE] f16
    _Float16*  XMUT = (_Float16*)(ws + alloc(SZH));
    float*     SKIP = (float*)(ws + alloc(SZF));
    float*     ATTL = (float*)(ws + alloc(SZF));
    float*     ATTU = (float*)(ws + alloc(SZF));
    float*     ASL0 = (float*)(ws + alloc(SZV));
    float*     ASL1 = (float*)(ws + alloc(SZV));
    float*     ASU0 = (float*)(ws + alloc(SZV));
    float*     ASU1 = (float*)(ws + alloc(SZV));
    float*     ADL  = (float*)(ws + alloc(2 * SZV));
    float*     ADU  = (float*)(ws + alloc(2 * SZV));
    float*     MRL  = (float*)(ws + alloc(2 * SZV));
    float*     MRU  = (float*)(ws + alloc(2 * SZV));
    float*     Z    = (float*)(ws + alloc(SZV));
    _Float16*  WLIFTT = (_Float16*)(ws + alloc(128 * 64 * sizeof(_Float16)));
    _Float16*  WT[2][3];
    for (int l = 0; l < 2; ++l)
        for (int k = 0; k < 3; ++k)
            WT[l][k] = (_Float16*)(ws + alloc(128 * 128 * sizeof(_Float16)));

    // weight transposes (f32 [K,N] -> f16 [N,K])
    k_t16<<<dim3(8, 4), 256, 0, stream>>>(Wlift, WLIFTT, 128, 64);
    for (int l = 0; l < 2; ++l)
        for (int k = 0; k < 3; ++k)
            k_t16<<<dim3(8, 8), 256, 0, stream>>>(W[l][k], WT[l][k], 128, 128);

    // lift + concat with x_1
    k_lift<<<NE / 16, 32, 0, stream>>>(x_0, eidx, WLIFTT, XA);
    k_copy_x1<<<(NE * C0 + 255) / 256, 256, 0, stream>>>(x_1, XA);

    dim3 gG(NE / 16, DH / 16);
    dim3 gT(NE / 16, DH / 16);
    const int cvB = (int)((NF + 255) / 256);
    float* Xin = XA;
    float* Xout = XB;
    for (int l = 0; l < 2; ++l) {
        k_cvt_f16<<<cvB, 256, 0, stream>>>(Xin, XH, (int)NF);
        k_gemm<<<gG, 32, 0, stream>>>(XH, WT[l][0], XML, NE, DH, DH);
        k_gemm<<<gG, 32, 0, stream>>>(XH, WT[l][1], XMU, NE, DH, DH);
        k_gemm<<<gG, 32, 0, stream>>>(XH, WT[l][2], SKIP, NE, DH, DH);
        k_t16<<<gT, 256, 0, stream>>>(XML, XMLT, NE, DH);
        k_t16<<<gT, 256, 0, stream>>>(XMU, XMUT, NE, DH);
        k_logits<<<NE / 256, 256, 0, stream>>>(XML, aS[l][0], aD[l][0], ASL0, ASL1, ADL);
        k_logits<<<NE / 256, 256, 0, stream>>>(XMU, aS[l][1], aD[l][1], ASU0, ASU1, ADU);
        k_rowmax<<<NE, 256, 0, stream>>>(mL, ASL0, ASL1, ADL, MRL);
        k_rowmax<<<NE, 256, 0, stream>>>(mU, ASU0, ASU1, ADU, MRU);
        k_attn<<<NE / 16, 32, 0, stream>>>(mL, ASL0, ASL1, ADL, MRL, XMLT, ATTL);
        k_attn<<<NE / 16, 32, 0, stream>>>(mU, ASU0, ASU1, ADU, MRU, XMUT, ATTU);
        k_combine<<<cvB, 256, 0, stream>>>(ATTL, ATTU, SKIP, Xout, (int)NF);
        float* t = Xin; Xin = Xout; Xout = t;   // layer2 reads XB, writes XA
    }
    // Xin now points at the final features (XA after two swaps)
    k_pool<<<NE / 256, 256, 0, stream>>>(Xin, att_pool, Z);
    k_topk<<<NE, 128, 0, stream>>>(Z, Xin, out);
}
